// HiroLRANInverse_73581379715575
// MI455X (gfx1250) — matrix operand order repaired and checked by reading, not compiled, not run on hardware
//
#include <hip/hip_runtime.h>

// =====================================================================
// HiroLRANInverse for MI455X (gfx1250, wave32, WMMA).
//
// Decomposition: encoder only matters at t=0 (tiny, f32 WMMA);
// Bu GEMM 65536x512x256 (~17 GF) + decoder 3x 65536x512x512 (~103 GF)
// run on V_WMMA_F32_16X16X32_BF16 via split-bf16 (hi/lo) arithmetic
// (3 wmma per K=32, ~1e-5 relative error, f32 accumulation);
// inv(W+eps I)^T via Neumann fixed point X <- I - E^T X (||E||~0.045,
// 10 iters -> ~1e-13) in exact f32 WMMA; scan is streaming (~12us).
// Total HBM traffic ~0.9 GB (~40us); compute-bound on the bf16 pipe.
// Workspace use: ~148 MB.
// =====================================================================

typedef __attribute__((ext_vector_type(2)))  float  v2f;
typedef __attribute__((ext_vector_type(4)))  float  v4f;
typedef __attribute__((ext_vector_type(8)))  float  v8f;
typedef __attribute__((ext_vector_type(8)))  __bf16 v8bf;
typedef __attribute__((ext_vector_type(16))) __bf16 v16bf;

#define WMMA_F32X4(a, b, c) \
  __builtin_amdgcn_wmma_f32_16x16x4_f32(false, (a), false, (b), (short)0, (c), false, false)
#define WMMA_BF16(a, b, c) \
  __builtin_amdgcn_wmma_f32_16x16x32_bf16(false, (a), false, (b), (short)0, (c), false, false)

#define DD 512
#define ACTN 256
#define BB 32
#define TT 2048
#define ROWS ((long)BB * TT)   // 65536
#define SLOPE 0.1f
#define EPSV 1e-6f

// Round-to-nearest-even split of f32 into bf16 hi + bf16 lo residual.
struct bf2 { __bf16 hi, lo; };
__device__ __forceinline__ bf2 split_bf16(float v) {
  bf2 s;
  unsigned u = __builtin_bit_cast(unsigned, v);
  unsigned r = (u + 0x7FFFu + ((u >> 16) & 1u)) & 0xFFFF0000u;
  s.hi = __builtin_bit_cast(__bf16, (unsigned short)(r >> 16));
  float rem = v - __builtin_bit_cast(float, r);
  unsigned u2 = __builtin_bit_cast(unsigned, rem);
  s.lo = __builtin_bit_cast(__bf16, (unsigned short)((u2 + 0x7FFFu + ((u2 >> 16) & 1u)) >> 16));
  return s;
}

// ---------------------------------------------------------------------
// Big GEMM, split-bf16: C[M,N] = op(A)[M,K] * B[K,N] (all f32 in memory).
// Block tile 128x64, 8 waves -> 32x32 patch each (2x2 of 16x16), BK=32.
// A in LDS m-major (hi/lo), B in LDS n-major/transposed (hi/lo) so both
// fragments are ds_load_b128 pairs matching the 16-bit WMMA layouts.
// act==1 fuses inverse-leaky-ReLU (x<0 -> 10x) into the A tile load.
// Requires: M%128==0, N%64==0, K%32==0.
// ---------------------------------------------------------------------
#define BM 128
#define BN 64
#define BK 32
#define A_PITCH 40   // bf16 per A-tile row (32 + pad); 40*2B % 16 == 0
#define B_PITCH 40   // bf16 per Bt-tile row

__global__ __launch_bounds__(256) void gemm_big_bf16_kernel(
    const float* __restrict__ A, long lda,
    const float* __restrict__ B, long ldb,
    float* __restrict__ C, long ldc, int K, int act) {
  __shared__ __bf16 Ah[2][BM * A_PITCH];
  __shared__ __bf16 Al[2][BM * A_PITCH];
  __shared__ __bf16 Bh[2][BN * B_PITCH];
  __shared__ __bf16 Bl[2][BN * B_PITCH];

  const int tid  = threadIdx.x;
  const int lane = tid & 31;
  const int wave = tid >> 5;      // 0..7
  const int wm   = wave & 3;      // 4 waves along M (32 rows each)
  const int wn   = wave >> 2;     // 2 waves along N (32 cols each)
  const long m0  = (long)blockIdx.y * BM;
  const long n0  = (long)blockIdx.x * BN;

  v8f acc[2][2];
#pragma unroll
  for (int i = 0; i < 2; ++i)
#pragma unroll
    for (int j = 0; j < 2; ++j) acc[i][j] = (v8f){0, 0, 0, 0, 0, 0, 0, 0};

  // A loader: thread -> (row am = tid&127, k = (tid>>7)*16 .. +15)
  const int am = tid & 127;
  const int ak = (tid >> 7) * 16;
  // B loader: thread -> (col bn = tid&63, k = (tid>>6)*8 .. +7), transposing.
  const int bn = tid & 63;
  const int bkk = (tid >> 6) * 8;

  auto load_tiles = [&](int buf, int kt) {
    const long kbase = (long)kt * BK;
    // ---- A tile: 16 consecutive k per thread, split, pack, 2x b128/array ----
    {
      const float* ap = A + (m0 + am) * lda + kbase + ak;
      v4f f[4];
#pragma unroll
      for (int q = 0; q < 4; ++q) f[q] = *(const v4f*)(ap + 4 * q);
      v8bf h0, h1, l0, l1;
#pragma unroll
      for (int j = 0; j < 8; ++j) {
        float v = f[j >> 2][j & 3];
        if (act) v = (v < 0.f) ? v * 10.0f : v;
        bf2 s = split_bf16(v);
        h0[j] = s.hi;
        l0[j] = s.lo;
      }
#pragma unroll
      for (int j = 0; j < 8; ++j) {
        float v = f[2 + (j >> 2)][j & 3];
        if (act) v = (v < 0.f) ? v * 10.0f : v;
        bf2 s = split_bf16(v);
        h1[j] = s.hi;
        l1[j] = s.lo;
      }
      const int o = am * A_PITCH + ak;
      *(v8bf*)&Ah[buf][o]     = h0;
      *(v8bf*)&Ah[buf][o + 8] = h1;
      *(v8bf*)&Al[buf][o]     = l0;
      *(v8bf*)&Al[buf][o + 8] = l1;
    }
    // ---- B tile: 8 k (stride ldb, wave-coalesced over n), transpose ----
    {
      const float* bp = B + (kbase + bkk) * ldb + n0 + bn;
      v8bf h, l;
#pragma unroll
      for (int j = 0; j < 8; ++j) {
        bf2 s = split_bf16(bp[(long)j * ldb]);
        h[j] = s.hi;
        l[j] = s.lo;
      }
      const int o = bn * B_PITCH + bkk;
      *(v8bf*)&Bh[buf][o] = h;
      *(v8bf*)&Bl[buf][o] = l;
    }
  };

  const int nk = K / BK;
  load_tiles(0, 0);
  __syncthreads();

  // Fragment addressing per the ISA 16-bit layouts:
  // A (16x32): lanes 0-15 hold K 0-7 & 16-23; lanes 16-31 hold K 8-15 & 24-31.
  // B (32x16): lanes 0-15 hold K 0-7 & 16-23; lanes 16-31 hold K 8-15 & 24-31.
  const int mrow = lane & 15;
  const int half = lane >> 4;

  for (int kt = 0; kt < nk; ++kt) {
    const int cur = kt & 1;
    if (kt + 1 < nk) load_tiles(cur ^ 1, kt + 1);

    v16bf afh[2], afl[2], bfh[2], bfl[2];
#pragma unroll
    for (int i = 0; i < 2; ++i) {
      const int o = (wm * 32 + i * 16 + mrow) * A_PITCH + 8 * half;
      v8bf p0 = *(const v8bf*)&Ah[cur][o];
      v8bf p1 = *(const v8bf*)&Ah[cur][o + 16];
      afh[i] = __builtin_shufflevector(p0, p1, 0, 1, 2, 3, 4, 5, 6, 7,
                                       8, 9, 10, 11, 12, 13, 14, 15);
      v8bf q0 = *(const v8bf*)&Al[cur][o];
      v8bf q1 = *(const v8bf*)&Al[cur][o + 16];
      afl[i] = __builtin_shufflevector(q0, q1, 0, 1, 2, 3, 4, 5, 6, 7,
                                       8, 9, 10, 11, 12, 13, 14, 15);
    }
#pragma unroll
    for (int j = 0; j < 2; ++j) {
      const int o = (wn * 32 + j * 16 + mrow) * B_PITCH + 8 * half;
      v8bf p0 = *(const v8bf*)&Bh[cur][o];
      v8bf p1 = *(const v8bf*)&Bh[cur][o + 16];
      bfh[j] = __builtin_shufflevector(p0, p1, 0, 1, 2, 3, 4, 5, 6, 7,
                                       8, 9, 10, 11, 12, 13, 14, 15);
      v8bf q0 = *(const v8bf*)&Bl[cur][o];
      v8bf q1 = *(const v8bf*)&Bl[cur][o + 16];
      bfl[j] = __builtin_shufflevector(q0, q1, 0, 1, 2, 3, 4, 5, 6, 7,
                                       8, 9, 10, 11, 12, 13, 14, 15);
    }

#pragma unroll
    for (int i = 0; i < 2; ++i)
#pragma unroll
      for (int j = 0; j < 2; ++j) {
        acc[i][j] = WMMA_BF16(afh[i], bfh[j], acc[i][j]);  // hi*hi
        acc[i][j] = WMMA_BF16(afh[i], bfl[j], acc[i][j]);  // hi*lo
        acc[i][j] = WMMA_BF16(afl[i], bfh[j], acc[i][j]);  // lo*hi
      }
    __syncthreads();
  }

  // C/D layout: VGPR r: lanes 0-15 = (M=r, N=lane); lanes 16-31 = (M=r+8,...).
#pragma unroll
  for (int i = 0; i < 2; ++i)
#pragma unroll
    for (int j = 0; j < 2; ++j) {
      const long crow = m0 + wm * 32 + i * 16 + (lane >> 4) * 8;
      const long ccol = n0 + wn * 32 + j * 16 + (lane & 15);
      float* cp = C + crow * ldc + ccol;
#pragma unroll
      for (int r = 0; r < 8; ++r) cp[(long)r * ldc] = acc[i][j][r];
    }
}

// ---------------------------------------------------------------------
// Small exact-f32 GEMM (V_WMMA_F32_16X16X4_F32): one wave = one 16x16
// tile, global loads. mode: 0 = acc(+bias); 1 = leaky-relu(acc+bias);
// 2 = I - acc (Neumann). Requires M%16==0, N%16==0, K%4==0.
// ---------------------------------------------------------------------
__global__ __launch_bounds__(256) void gemm_small_kernel(
    const float* __restrict__ A, long lda,
    const float* __restrict__ B, long ldb,
    const float* __restrict__ bias,
    float* __restrict__ C, long ldc,
    int K, int Ntiles, int totalTiles, int mode) {
  const int gwave = (int)((blockIdx.x * blockDim.x + threadIdx.x) >> 5);
  if (gwave >= totalTiles) return;  // whole-wave uniform exit (EXEC stays full)
  const int lane = threadIdx.x & 31;
  const long m0 = (long)(gwave / Ntiles) * 16;
  const long n0 = (long)(gwave % Ntiles) * 16;

  const int mrow  = lane & 15;
  const int khalf = (lane >> 4) * 2;

  v8f acc = (v8f){0, 0, 0, 0, 0, 0, 0, 0};
  for (int k0 = 0; k0 < K; k0 += 4) {
    v2f a, b;
    const float* ap = A + (m0 + mrow) * lda + k0 + khalf;
    a.x = ap[0];
    a.y = ap[1];
    b.x = B[(long)(k0 + khalf + 0) * ldb + n0 + mrow];
    b.y = B[(long)(k0 + khalf + 1) * ldb + n0 + mrow];
    acc = WMMA_F32X4(a, b, acc);
  }

  const long row0 = m0 + ((lane >> 4) << 3);
  const long col  = n0 + (lane & 15);
#pragma unroll
  for (int r = 0; r < 8; ++r) {
    const long row = row0 + r;
    float v = acc[r];
    if (mode == 2) {
      v = ((row == col) ? 1.0f : 0.0f) - v;
    } else {
      if (bias) v += bias[col];
      if (mode == 1) v = (v < 0.f) ? SLOPE * v : v;
    }
    C[row * ldc + col] = v;
  }
}

// out[j][i] = in[i][j] + diag_add * (i==j); in is R x C row-major.
__global__ void transpose_kernel(const float* __restrict__ in, int R, int C,
                                 float* __restrict__ out, float diag_add) {
  const long idx = (long)blockIdx.x * blockDim.x + threadIdx.x;
  if (idx >= (long)R * C) return;
  const int i = (int)(idx / C), j = (int)(idx % C);
  float v = in[(long)i * C + j];
  if (i == j) v += diag_add;
  out[(long)j * R + i] = v;
}

__global__ void set_identity_kernel(float* __restrict__ X, int n) {
  const long idx = (long)blockIdx.x * blockDim.x + threadIdx.x;
  if (idx >= (long)n * n) return;
  X[idx] = ((idx / n) == (idx % n)) ? 1.0f : 0.0f;
}

// x0[b][d] = padded_input[b][0][d]
__global__ void gather_x0_kernel(const float* __restrict__ pin, float* __restrict__ x0) {
  const int idx = blockIdx.x * blockDim.x + threadIdx.x;  // 32*512
  if (idx >= BB * DD) return;
  const int b = idx >> 9, d = idx & 511;
  x0[idx] = pin[(long)b * TT * (2 * DD) + d];
}

// In-place diagonal scan over t: z[t] = z[t-1]*a + Bu[t], z[-1] = z0.
__global__ void scan_kernel(float* __restrict__ bu, const float* __restrict__ z0,
                            const float* __restrict__ A_diag) {
  const int idx = blockIdx.x * blockDim.x + threadIdx.x;  // 32*512
  if (idx >= BB * DD) return;
  const int b = idx >> 9, d = idx & 511;
  float a = A_diag[d];
  a = fminf(0.95f, fmaxf(-0.95f, a));
  float carry = z0[idx];
  float* p = bu + (long)b * TT * DD + d;
  float nxt = p[0];
  for (int t = 0; t < TT; ++t) {
    const float cur = nxt;
    if (t + 1 < TT) nxt = p[(long)(t + 1) * DD];  // prefetch next before store
    const float v = fmaf(carry, a, cur);
    p[(long)t * DD] = v;
    carry = v;
  }
}

// ---------------------------------------------------------------------
extern "C" void kernel_launch(void* const* d_in, const int* in_sizes, int n_in,
                              void* d_out, int out_size, void* d_ws, size_t ws_size,
                              hipStream_t stream) {
  const float* padded = (const float*)d_in[0];
  const float* W[3]   = {(const float*)d_in[1], (const float*)d_in[3], (const float*)d_in[5]};
  const float* bia[3] = {(const float*)d_in[2], (const float*)d_in[4], (const float*)d_in[6]};
  const float* A_diag = (const float*)d_in[7];
  const float* B_w    = (const float*)d_in[8];
  float* out = (float*)d_out;

  // Workspace layout (floats); ~148 MB total.
  float* ws = (float*)d_ws;
  size_t off = 0;
  auto wsalloc = [&](size_t n) {
    float* p = ws + off;
    off += (n + 15) & ~(size_t)15;  // keep 64B alignment
    return p;
  };
  float* buf0 = wsalloc(ROWS * DD);  // Bu -> z1 (in place) -> decoder ping-pong
  float *Wt[3], *Et[3], *Xa[3], *Xb[3];
  for (int k = 0; k < 3; ++k) {
    Wt[k] = wsalloc(DD * DD);
    Et[k] = wsalloc(DD * DD);
    Xa[k] = wsalloc(DD * DD);
    Xb[k] = wsalloc(DD * DD);
  }
  float* Bwt = wsalloc(ACTN * DD);
  float* z0a = wsalloc(BB * DD);
  float* z0b = wsalloc(BB * DD);

  // ---- Prep: Wt (encoder), Et = (W + eps I - I)^T (Neumann), Bwt ----
  for (int k = 0; k < 3; ++k) {
    transpose_kernel<<<(DD * DD + 255) / 256, 256, 0, stream>>>(W[k], DD, DD, Wt[k], 0.0f);
    transpose_kernel<<<(DD * DD + 255) / 256, 256, 0, stream>>>(W[k], DD, DD, Et[k], EPSV - 1.0f);
    set_identity_kernel<<<(DD * DD + 255) / 256, 256, 0, stream>>>(Xa[k], DD);
  }
  transpose_kernel<<<(DD * ACTN + 255) / 256, 256, 0, stream>>>(B_w, DD, ACTN, Bwt, 0.0f);

  // ---- Neumann: X <- I - Et @ X (10 iters) -> X = inv(M)^T, exact f32 ----
  float* Xf[3];
  {
    const int tiles = (DD / 16) * (DD / 16);      // 1024
    const int blocks = (tiles * 32 + 255) / 256;  // 128
    for (int k = 0; k < 3; ++k) {
      float* cur = Xa[k];
      float* nxt = Xb[k];
      for (int it = 0; it < 10; ++it) {
        gemm_small_kernel<<<blocks, 256, 0, stream>>>(Et[k], DD, cur, DD, nullptr,
                                                      nxt, DD, DD, DD / 16, tiles, 2);
        float* t = cur; cur = nxt; nxt = t;
      }
      Xf[k] = cur;
    }
  }

  // ---- Encoder (t=0 rows only): z = lrelu(z @ W^T + b), 3 layers ----
  gather_x0_kernel<<<(BB * DD + 255) / 256, 256, 0, stream>>>(padded, z0a);
  {
    const int tiles = (BB / 16) * (DD / 16);      // 64
    const int blocks = (tiles * 32 + 255) / 256;  // 8
    gemm_small_kernel<<<blocks, 256, 0, stream>>>(z0a, DD, Wt[0], DD, bia[0], z0b, DD, DD, DD / 16, tiles, 1);
    gemm_small_kernel<<<blocks, 256, 0, stream>>>(z0b, DD, Wt[1], DD, bia[1], z0a, DD, DD, DD / 16, tiles, 1);
    gemm_small_kernel<<<blocks, 256, 0, stream>>>(z0a, DD, Wt[2], DD, bia[2], z0b, DD, DD, DD / 16, tiles, 1);
  }

  // ---- Bu = u_t1 @ B_w^T (A rows live inside padded_input, lda=1024) ----
  {
    dim3 grid(DD / BN, (unsigned)(ROWS / BM));  // (8, 512)
    gemm_big_bf16_kernel<<<grid, 256, 0, stream>>>(padded + (DD + ACTN), 2 * DD,
                                                   Bwt, DD, buf0, DD, ACTN, 0);
  }

  // ---- Diagonal scan, in place over buf0 ----
  scan_kernel<<<(BB * DD + 255) / 256, 256, 0, stream>>>(buf0, z0b, A_diag);

  // ---- Decoder: x = inv_lrelu(x) @ inv(W+eps I)^T, layers w2, w1, w0 ----
  {
    dim3 grid(DD / BN, (unsigned)(ROWS / BM));  // (8, 512)
    gemm_big_bf16_kernel<<<grid, 256, 0, stream>>>(buf0, DD, Xf[2], DD, out, DD, DD, 1);
    gemm_big_bf16_kernel<<<grid, 256, 0, stream>>>(out, DD, Xf[1], DD, buf0, DD, DD, 1);
    gemm_big_bf16_kernel<<<grid, 256, 0, stream>>>(buf0, DD, Xf[0], DD, out, DD, DD, 1);
  }
}
